// ConformerEncoderLayer_47579647705803
// MI455X (gfx1250) — compile-verified, matrix operand order checked by reading, not run on hardware
//
#include <hip/hip_runtime.h>

typedef __attribute__((ext_vector_type(16))) __bf16 v16bf;
typedef __attribute__((ext_vector_type(8)))  __bf16 v8bf;
typedef __attribute__((ext_vector_type(8)))  float  v8f;

__device__ __forceinline__ unsigned bf16bits(float f) {     // RNE f32 -> bf16 bits
  union { float f; unsigned u; } v; v.f = f;
  return (v.u + 0x7FFFu + ((v.u >> 16) & 1u)) >> 16;
}
__device__ __forceinline__ unsigned pack2bf(float lo, float hi) {
  union { float f; unsigned u; } v; v.f = hi;
  unsigned h = (v.u + 0x7FFFu + ((v.u >> 16) & 1u)) & 0xFFFF0000u;
  return h | bf16bits(lo);
}
__device__ __forceinline__ __bf16 f2bf(float f) {
  union { unsigned short s; __bf16 b; } o; o.s = (unsigned short)bf16bits(f);
  return o.b;
}

__device__ __forceinline__ float dswish(float v) {           // x * sigmoid(x-1)
  return v / (1.0f + __expf(1.0f - v));
}

// Build a v16bf A/B fragment for v_wmma_f32_16x16x32_bf16 from a row-major row.
// Per ISA: lane kb = (lane>=16)?8:0 ; element e<8 -> K=kb+e ; e>=8 -> K=16+kb+(e-8).
__device__ __forceinline__ v16bf frag16(const __bf16* row, int kb) {
  v8bf lo = *reinterpret_cast<const v8bf*>(row + kb);
  v8bf hi = *reinterpret_cast<const v8bf*>(row + 16 + kb);
  v16bf f;
#pragma unroll
  for (int i = 0; i < 8; ++i) { f[i] = lo[i]; f[i + 8] = hi[i]; }
  return f;
}

// ---------------------------------------------------------------------------
// Generic WMMA GEMM:  C(M,N) = act(A(M,K) * W(N,K)^T + bias) [+ resid]
// block = 256 threads = 8 waves; block tile 64x64; wave tile 16x32 (A-frag
// reused across two WMMAs); K step 32. N, K multiples of 64/32; M guarded.
// ---------------------------------------------------------------------------
__global__ void __launch_bounds__(256)
gemm_bf16(const float* __restrict__ A, const float* __restrict__ W,
          const float* __restrict__ bias, const float* __restrict__ resid,
          float* __restrict__ C, int M, int N, int K, int act)
{
  __shared__ __align__(16) __bf16 As[64][40];   // 80B row stride: 16B aligned frags
  __shared__ __align__(16) __bf16 Bs[64][40];

  const int tid  = threadIdx.x;
  const int lane = tid & 31;
  const int wave = tid >> 5;
  const int bm   = blockIdx.y * 64;
  const int bn   = blockIdx.x * 64;
  const int tm0  = (wave >> 1) * 16;            // 0,16,32,48
  const int tn0  = (wave & 1) * 32;             // 0,32
  const int l15  = lane & 15;
  const int kb   = (lane >> 4) << 3;            // 0 or 8
  const int ihi  = kb;                          // D-layout row offset

  // staging: thread loads 8 contiguous f32 from one row of A and one row of W
  const int  sr   = tid >> 2;                   // 0..63
  const int  sc   = (tid & 3) * 8;              // 0,8,16,24
  const int  agr  = bm + sr;
  const bool aval = (agr < M);
  const float* aptr = A + (size_t)agr * K + sc;
  const float* wptr = W + (size_t)(bn + sr) * K + sc;

  v8f acc0 = {0.f,0.f,0.f,0.f,0.f,0.f,0.f,0.f};
  v8f acc1 = {0.f,0.f,0.f,0.f,0.f,0.f,0.f,0.f};

  for (int k0 = 0; k0 < K; k0 += 32) {
    if (k0 + 32 < K) {                          // prefetch next K tile
      if (aval) __builtin_prefetch(aptr + k0 + 32, 0, 1);
      __builtin_prefetch(wptr + k0 + 32, 0, 1);
    }
    float4 a0, a1;
    if (aval) {
      const float4* p = (const float4*)(aptr + k0);
      a0 = p[0]; a1 = p[1];
    } else {
      a0 = make_float4(0.f, 0.f, 0.f, 0.f); a1 = a0;
    }
    const float4* q = (const float4*)(wptr + k0);
    float4 w0 = q[0], w1 = q[1];

    *(uint4*)&As[sr][sc] = make_uint4(pack2bf(a0.x, a0.y), pack2bf(a0.z, a0.w),
                                      pack2bf(a1.x, a1.y), pack2bf(a1.z, a1.w));
    *(uint4*)&Bs[sr][sc] = make_uint4(pack2bf(w0.x, w0.y), pack2bf(w0.z, w0.w),
                                      pack2bf(w1.x, w1.y), pack2bf(w1.z, w1.w));
    __syncthreads();

    v16bf af  = frag16(&As[tm0 + l15][0], kb);
    v16bf bf0 = frag16(&Bs[tn0 + l15][0], kb);
    v16bf bf1 = frag16(&Bs[tn0 + 16 + l15][0], kb);
    acc0 = __builtin_amdgcn_wmma_f32_16x16x32_bf16(false, af, false, bf0,
                                                   (short)0, acc0, false, false);
    acc1 = __builtin_amdgcn_wmma_f32_16x16x32_bf16(false, af, false, bf1,
                                                   (short)0, acc1, false, false);
    __syncthreads();
  }

  const int n0 = bn + tn0 + l15;
#pragma unroll
  for (int r = 0; r < 8; ++r) {
    int m = bm + tm0 + r + ihi;
    if (m < M) {
      size_t base = (size_t)m * N;
      float v0 = acc0[r] + (bias ? bias[n0] : 0.0f);
      float v1 = acc1[r] + (bias ? bias[n0 + 16] : 0.0f);
      if (act) { v0 = dswish(v0); v1 = dswish(v1); }
      if (resid) { v0 += resid[base + n0]; v1 += resid[base + n0 + 16]; }
      C[base + n0]      = v0;
      C[base + n0 + 16] = v1;
    }
  }
}

// ---------------------------------------------------------------------------
// f32 -> bf16 conversion (for p = pos_emb @ w_pos^T)
// ---------------------------------------------------------------------------
__global__ void __launch_bounds__(256)
cvt_f32_bf16(const float* __restrict__ in, unsigned* __restrict__ out, int npairs)
{
  int i = blockIdx.x * 256 + threadIdx.x;
  if (i < npairs) out[i] = pack2bf(in[2 * i], in[2 * i + 1]);
}

// ---------------------------------------------------------------------------
// Pack q+u, q+v, k, v into bf16 (N,H,L,d) layout. idx over N*H*L*16 pairs
// ---------------------------------------------------------------------------
__global__ void __launch_bounds__(256)
pack_qkv(const float* __restrict__ qkv, const float* __restrict__ pbu,
         const float* __restrict__ pbv, unsigned* __restrict__ qu,
         unsigned* __restrict__ qv, unsigned* __restrict__ kk, unsigned* __restrict__ vv,
         int L, int NB, int H)
{
  size_t idx = (size_t)blockIdx.x * 256 + threadIdx.x;   // pair index (d/2 = 16 per (n,h,l))
  int dp = (int)(idx & 15);                              // pair within head dim
  size_t t = idx >> 4;
  int l = (int)(t % L); t /= L;
  int h = (int)(t % H);
  int n = (int)(t / H);
  int dd = dp * 2;
  const float* row = qkv + ((size_t)l * NB + n) * 768 + h * 32 + dd;
  float q0 = row[0],   q1 = row[1];
  float k0 = row[256], k1 = row[257];
  float v0 = row[512], v1 = row[513];
  float u0 = pbu[h * 32 + dd], u1 = pbu[h * 32 + dd + 1];
  float w0 = pbv[h * 32 + dd], w1 = pbv[h * 32 + dd + 1];
  qu[idx] = pack2bf(q0 + u0, q1 + u1);
  qv[idx] = pack2bf(q0 + w0, q1 + w1);
  kk[idx] = pack2bf(k0, k1);
  vv[idx] = pack2bf(v0, v1);
}

// ---------------------------------------------------------------------------
// scores_in projection: sp[n,g,i,j] = sum_h asin[n,i,j,h] * proj_in[h,g]
// ---------------------------------------------------------------------------
__global__ void __launch_bounds__(256)
proj_scores_in(const float* __restrict__ asin, const float* __restrict__ proj,
               float* __restrict__ sp, int L)
{
  __shared__ float P[64];
  if (threadIdx.x < 64) P[threadIdx.x] = proj[threadIdx.x];
  __syncthreads();
  size_t idx = (size_t)blockIdx.x * 256 + threadIdx.x;   // over N*L*L (exact)
  size_t LL = (size_t)L * L;
  size_t n = idx / LL, ij = idx % LL;
  const float* ap = asin + idx * 8;
  float a[8];
#pragma unroll
  for (int h = 0; h < 8; ++h) a[h] = ap[h];
#pragma unroll
  for (int g = 0; g < 8; ++g) {
    float s = 0.f;
#pragma unroll
    for (int h = 0; h < 8; ++h) s += a[h] * P[h * 8 + g];
    sp[(n * 8 + g) * LL + ij] = s;
  }
}

// ---------------------------------------------------------------------------
// total[n,h,i,j] = (ac + bd_shifted)*scaling + sp[n,h,i,j]
// ac  : (q+u)·k^T        -> 1 WMMA per 16x16 tile (K = d = 32)
// bd  : (q+v)·p^T with rel-shift folded in: needed p rows form a 31-row band
//       -> 2 WMMAs into 16x31 band in LDS, then shifted gather.
// grid: (L/128, L/16, N*H), block 256 (8 waves, wave = one 16x16 j-tile)
// ---------------------------------------------------------------------------
__global__ void __launch_bounds__(256)
attn_total(const __bf16* __restrict__ qu, const __bf16* __restrict__ qv,
           const __bf16* __restrict__ kk, const __bf16* __restrict__ pbf,
           const float* __restrict__ sp, float* __restrict__ total,
           int L, int H, float scaling)
{
  __shared__ float band[8][16][33];
  const int tid  = threadIdx.x;
  const int lane = tid & 31;
  const int wave = tid >> 5;
  const int nh = blockIdx.z;
  const int h  = nh % H;
  const int i0 = blockIdx.y * 16;
  const int j0 = blockIdx.x * 128 + wave * 16;
  const int l15 = lane & 15;
  const int kb  = (lane >> 4) << 3;
  const int ihi = kb;

  const __bf16* qub = qu + (size_t)nh * L * 32;
  const __bf16* qvb = qv + (size_t)nh * L * 32;
  const __bf16* kkb = kk + (size_t)nh * L * 32;

  v16bf aqu = frag16(qub + (size_t)(i0 + l15) * 32, kb);
  v16bf aqv = frag16(qvb + (size_t)(i0 + l15) * 32, kb);
  v16bf bk  = frag16(kkb + (size_t)(j0 + l15) * 32, kb);

  v8f zero = {0.f,0.f,0.f,0.f,0.f,0.f,0.f,0.f};
  v8f ac = __builtin_amdgcn_wmma_f32_16x16x32_bf16(false, aqu, false, bk,
                                                   (short)0, zero, false, false);

  // band of p rows: m = (L-1) - i + j, tile band base:
  const int mmax  = 2 * L - 2;
  const int mbase = (L - 1) - (i0 + 15) + j0;            // >= 0
  int r1 = mbase + l15;
  int r2 = mbase + 16 + l15; if (r2 > mmax) r2 = mmax;   // j'=31 never gathered
  v16bf bp1 = frag16(pbf + (size_t)r1 * 256 + h * 32, kb);
  v16bf bp2 = frag16(pbf + (size_t)r2 * 256 + h * 32, kb);
  v8f bd1 = __builtin_amdgcn_wmma_f32_16x16x32_bf16(false, aqv, false, bp1,
                                                    (short)0, zero, false, false);
  v8f bd2 = __builtin_amdgcn_wmma_f32_16x16x32_bf16(false, aqv, false, bp2,
                                                    (short)0, zero, false, false);

  // D layout: element r -> (i_loc = r+ihi, col = l15)
#pragma unroll
  for (int r = 0; r < 8; ++r) {
    band[wave][r + ihi][l15] = bd1[r];
    if (l15 < 15) band[wave][r + ihi][16 + l15] = bd2[r];   // j' 16..30
    else          band[wave][r + ihi][31]       = bd2[r];   // unused slot
  }
  __syncthreads();

  const float* spb = sp    + ((size_t)nh * L + i0) * L + j0;
  float*       tob = total + ((size_t)nh * L + i0) * L + j0;
#pragma unroll
  for (int r = 0; r < 8; ++r) {
    int i_loc = r + ihi;
    int jp = 15 - i_loc + l15;                 // 0..30
    float bd = band[wave][i_loc][jp];
    tob[(size_t)i_loc * L + l15] = (ac[r] + bd) * scaling + spb[(size_t)i_loc * L + l15];
  }
}

// ---------------------------------------------------------------------------
// scores_out[n,i,j,g] = sum_h total[n,h,i,j]*proj_out[h,g] + asin[n,i,j,g]
// ---------------------------------------------------------------------------
__global__ void __launch_bounds__(256)
scores_out_k(const float* __restrict__ total, const float* __restrict__ proj,
             const float* __restrict__ asin, float* __restrict__ out, int L)
{
  __shared__ float P[64];
  if (threadIdx.x < 64) P[threadIdx.x] = proj[threadIdx.x];
  __syncthreads();
  size_t idx = (size_t)blockIdx.x * 256 + threadIdx.x;   // over N*L*L (exact)
  size_t LL = (size_t)L * L;
  size_t n = idx / LL, ij = idx % LL;
  float t[8];
#pragma unroll
  for (int h = 0; h < 8; ++h) t[h] = total[(n * 8 + h) * LL + ij];
  const float* ap = asin + idx * 8;
  float* op = out + idx * 8;
#pragma unroll
  for (int g = 0; g < 8; ++g) {
    float s = 0.f;
#pragma unroll
    for (int h = 0; h < 8; ++h) s += t[h] * P[h * 8 + g];
    op[g] = s + ap[g];
  }
}

// ---------------------------------------------------------------------------
// in-place row softmax over last dim (L); one block per (n,h,i) row
// ---------------------------------------------------------------------------
__global__ void __launch_bounds__(256)
softmax_rows(float* __restrict__ t, int L)
{
  float* row = t + (size_t)blockIdx.x * L;
  __shared__ float red[256];
  int tid = threadIdx.x;
  float mx = -3.4e38f;
  for (int j = tid; j < L; j += 256) mx = fmaxf(mx, row[j]);
  red[tid] = mx; __syncthreads();
  for (int s = 128; s > 0; s >>= 1) { if (tid < s) red[tid] = fmaxf(red[tid], red[tid + s]); __syncthreads(); }
  mx = red[0]; __syncthreads();
  float sum = 0.f;
  for (int j = tid; j < L; j += 256) { float e = __expf(row[j] - mx); row[j] = e; sum += e; }
  red[tid] = sum; __syncthreads();
  for (int s = 128; s > 0; s >>= 1) { if (tid < s) red[tid] += red[tid + s]; __syncthreads(); }
  float inv = 1.0f / red[0];
  for (int j = tid; j < L; j += 256) row[j] *= inv;
}

// ---------------------------------------------------------------------------
// ctx = attn(L,L) @ V(L,32) per (n,h); output directly in (L, N, H*32) layout.
// grid: (L/128, N*H); block 256 (8 waves, wave = one 16-row M tile, both d-tiles)
// ---------------------------------------------------------------------------
__global__ void __launch_bounds__(256)
ctx_kernel(const float* __restrict__ attn, const __bf16* __restrict__ vv,
           float* __restrict__ ctxm, int L, int NB, int H)
{
  __shared__ __align__(16) __bf16 Vs[32][40];
  const int tid  = threadIdx.x;
  const int lane = tid & 31;
  const int wave = tid >> 5;
  const int nh = blockIdx.y;
  const int n  = nh / H;
  const int h  = nh % H;
  const int i0 = blockIdx.x * 128 + wave * 16;
  const int l15 = lane & 15;
  const int kb  = (lane >> 4) << 3;
  const int ihi = kb;

  const float*  ab = attn + (size_t)nh * L * L;
  const __bf16* vb = vv   + (size_t)nh * L * 32;

  v8f acc0 = {0.f,0.f,0.f,0.f,0.f,0.f,0.f,0.f};
  v8f acc1 = {0.f,0.f,0.f,0.f,0.f,0.f,0.f,0.f};

  for (int k0 = 0; k0 < L; k0 += 32) {
    { // stage V rows k0..k0+31 (bf16, shared by all 8 waves): uint2 = 4 bf16
      int r = tid >> 3;                 // 0..31
      int c = (tid & 7) * 4;            // 0..28
      *(uint2*)&Vs[r][c] = *(const uint2*)(vb + (size_t)(k0 + r) * 32 + c);
    }
    __syncthreads();

    const float* ap = ab + (size_t)(i0 + l15) * L + k0;
    float4 f0 = *(const float4*)(ap + kb);
    float4 f1 = *(const float4*)(ap + kb + 4);
    float4 f2 = *(const float4*)(ap + 16 + kb);
    float4 f3 = *(const float4*)(ap + 20 + kb);
    v16bf af;
    af[0]  = f2bf(f0.x); af[1]  = f2bf(f0.y); af[2]  = f2bf(f0.z); af[3]  = f2bf(f0.w);
    af[4]  = f2bf(f1.x); af[5]  = f2bf(f1.y); af[6]  = f2bf(f1.z); af[7]  = f2bf(f1.w);
    af[8]  = f2bf(f2.x); af[9]  = f2bf(f2.y); af[10] = f2bf(f2.z); af[11] = f2bf(f2.w);
    af[12] = f2bf(f3.x); af[13] = f2bf(f3.y); af[14] = f2bf(f3.z); af[15] = f2bf(f3.w);

    v16bf b0, b1;
#pragma unroll
    for (int e = 0; e < 8; ++e) {
      b0[e]     = Vs[kb + e][l15];      b1[e]     = Vs[kb + e][16 + l15];
      b0[e + 8] = Vs[16 + kb + e][l15]; b1[e + 8] = Vs[16 + kb + e][16 + l15];
    }
    acc0 = __builtin_amdgcn_wmma_f32_16x16x32_bf16(false, af, false, b0, (short)0, acc0, false, false);
    acc1 = __builtin_amdgcn_wmma_f32_16x16x32_bf16(false, af, false, b1, (short)0, acc1, false, false);
    __syncthreads();
  }

#pragma unroll
  for (int r = 0; r < 8; ++r) {
    int li = i0 + r + ihi;
    float* o = ctxm + ((size_t)li * NB + n) * (size_t)(H * 32) + h * 32;
    o[l15]      = acc0[r];
    o[16 + l15] = acc1[r];
  }
}

// ---------------------------------------------------------------------------
// GLU over channel halves: g[row,c] = h[row,c] * sigmoid(h[row,512+c])
// ---------------------------------------------------------------------------
__global__ void __launch_bounds__(256)
glu_kernel(const float* __restrict__ h, float* __restrict__ g)
{
  size_t idx = (size_t)blockIdx.x * 256 + threadIdx.x;   // over R*512 (exact)
  size_t row = idx >> 9;
  int c = (int)(idx & 511);
  float a = h[row * 1024 + c];
  float b = h[row * 1024 + 512 + c];
  g[idx] = a / (1.0f + __expf(-b));
}

// ---------------------------------------------------------------------------
// Depthwise conv (K=31, pad 15) over L, per (n,c); then DoubleSwish.
// layout: rows = l*NB + n (NB == 4), 512 channels
// ---------------------------------------------------------------------------
__global__ void __launch_bounds__(256)
dwconv_kernel(const float* __restrict__ g, const float* __restrict__ wdw,
              const float* __restrict__ bdw, float* __restrict__ o, int L)
{
  size_t idx = (size_t)blockIdx.x * 256 + threadIdx.x;   // over L*NB*512 (exact)
  int c = (int)(idx & 511);
  size_t row = idx >> 9;
  int n = (int)(row & 3);
  int l = (int)(row >> 2);
  float acc = bdw[c];
#pragma unroll
  for (int k = 0; k < 31; ++k) {
    int l2 = l + k - 15;
    if (l2 >= 0 && l2 < L)
      acc += g[((size_t)(l2 * 4 + n) << 9) + c] * wdw[c * 31 + k];
  }
  o[idx] = dswish(acc);
}

// ---------------------------------------------------------------------------
// BasicNorm: x *= rsqrt(mean(x^2) + exp(eps)); one block per row of 512
// ---------------------------------------------------------------------------
__global__ void __launch_bounds__(256)
basic_norm(const float* __restrict__ x, const float* __restrict__ eps,
           float* __restrict__ out, int E)
{
  const float* row = x + (size_t)blockIdx.x * E;
  __shared__ float red[256];
  int tid = threadIdx.x;
  float s = 0.f;
  for (int c = tid; c < E; c += 256) { float v = row[c]; s += v * v; }
  red[tid] = s; __syncthreads();
  for (int st = 128; st > 0; st >>= 1) { if (tid < st) red[tid] += red[tid + st]; __syncthreads(); }
  float scale = rsqrtf(red[0] / (float)E + __expf(eps[0]));
  float* o = out + (size_t)blockIdx.x * E;
  for (int c = tid; c < E; c += 256) o[c] = row[c] * scale;
}

// ===========================================================================
extern "C" void kernel_launch(void* const* d_in, const int* in_sizes, int n_in,
                              void* d_out, int out_size, void* d_ws, size_t ws_size,
                              hipStream_t stream)
{
  (void)in_sizes; (void)n_in; (void)out_size; (void)ws_size;

  const float* src      = (const float*)d_in[0];
  const float* pos_emb  = (const float*)d_in[1];
  const float* asin     = (const float*)d_in[2];
  const float* w_in     = (const float*)d_in[3];
  const float* b_in     = (const float*)d_in[4];
  const float* w_pos    = (const float*)d_in[5];
  const float* pbu      = (const float*)d_in[6];
  const float* pbv      = (const float*)d_in[7];
  const float* proj_in  = (const float*)d_in[8];
  const float* proj_out = (const float*)d_in[9];
  const float* w_out    = (const float*)d_in[10];
  const float* b_out    = (const float*)d_in[11];
  const float* w_ff1m   = (const float*)d_in[12];
  const float* b_ff1m   = (const float*)d_in[13];
  const float* w_ff2m   = (const float*)d_in[14];
  const float* b_ff2m   = (const float*)d_in[15];
  const float* w_ff1    = (const float*)d_in[16];
  const float* b_ff1    = (const float*)d_in[17];
  const float* w_ff2    = (const float*)d_in[18];
  const float* b_ff2    = (const float*)d_in[19];
  const float* w_pw1    = (const float*)d_in[20];
  const float* b_pw1    = (const float*)d_in[21];
  const float* w_dw     = (const float*)d_in[22];
  const float* b_dw     = (const float*)d_in[23];
  const float* w_pw2    = (const float*)d_in[24];
  const float* b_pw2    = (const float*)d_in[25];
  const float* neps     = (const float*)d_in[26];

  const int L = 1024, NB = 4, E = 512, H = 8, DFF = 2048;
  const int R = L * NB;                 // 4096 token rows

  char* ws = (char*)d_ws;
  size_t off = 0;
  auto alloc = [&](size_t bytes) { size_t o = off; off += (bytes + 255) & ~(size_t)255; return o; };
  float*  x    = (float*) (ws + alloc((size_t)R * E * 4));
  float*  h1   = (float*) (ws + alloc((size_t)R * DFF * 4));
  float*  qkv  = (float*) (ws + alloc((size_t)R * 768 * 4));
  float*  ctxm = (float*) (ws + alloc((size_t)R * 256 * 4));
  float*  glu  = (float*) (ws + alloc((size_t)R * E * 4));
  float*  dwo  = (float*) (ws + alloc((size_t)R * E * 4));
  float*  p32  = (float*) (ws + alloc((size_t)2047 * 256 * 4));
  __bf16* pbf  = (__bf16*)(ws + alloc((size_t)2047 * 256 * 2));
  __bf16* qu   = (__bf16*)(ws + alloc((size_t)NB * H * L * 32 * 2));
  __bf16* qvp  = (__bf16*)(ws + alloc((size_t)NB * H * L * 32 * 2));
  __bf16* kkp  = (__bf16*)(ws + alloc((size_t)NB * H * L * 32 * 2));
  __bf16* vvp  = (__bf16*)(ws + alloc((size_t)NB * H * L * 32 * 2));
  float*  sp   = (float*) (ws + alloc((size_t)NB * H * L * L * 4));
  float*  tot  = (float*) (ws + alloc((size_t)NB * H * L * L * 4));

  float* out_x = (float*)d_out;                          // (L,N,E)
  float* out_s = (float*)d_out + (size_t)L * NB * E;     // (N,L,L,H)

  const dim3 blk(256);
  const float scaling = 0.17677669529663687f;            // d^-0.5, d=32

  // macaron FFN: x = src + W2 * dswish(W1*src + b1) + b2
  gemm_bf16<<<dim3(DFF / 64, R / 64), blk, 0, stream>>>(src, w_ff1m, b_ff1m, nullptr, h1, R, DFF, E, 1);
  gemm_bf16<<<dim3(E / 64,  R / 64), blk, 0, stream>>>(h1, w_ff2m, b_ff2m, src, x, R, E, DFF, 0);

  // attention projections
  gemm_bf16<<<dim3(768 / 64, R / 64), blk, 0, stream>>>(x, w_in, b_in, nullptr, qkv, R, 768, E, 0);
  gemm_bf16<<<dim3(256 / 64, (2047 + 63) / 64), blk, 0, stream>>>(pos_emb, w_pos, nullptr, nullptr, p32, 2047, 256, E, 0);
  cvt_f32_bf16<<<(2047 * 128 + 255) / 256, blk, 0, stream>>>(p32, (unsigned*)pbf, 2047 * 128);
  pack_qkv<<<(NB * H * L * 16) / 256, blk, 0, stream>>>(qkv, pbu, pbv, (unsigned*)qu, (unsigned*)qvp,
                                                        (unsigned*)kkp, (unsigned*)vvp, L, NB, H);

  // scores
  proj_scores_in<<<(NB * L * L) / 256, blk, 0, stream>>>(asin, proj_in, sp, L);
  attn_total<<<dim3(L / 128, L / 16, NB * H), blk, 0, stream>>>(qu, qvp, kkp, pbf, sp, tot, L, H, scaling);
  scores_out_k<<<(NB * L * L) / 256, blk, 0, stream>>>(tot, proj_out, asin, out_s, L);

  // softmax -> context -> output projection (+ residual into x)
  softmax_rows<<<NB * H * L, blk, 0, stream>>>(tot, L);
  ctx_kernel<<<dim3(L / 128, NB * H), blk, 0, stream>>>(tot, vvp, ctxm, L, NB, H);
  gemm_bf16<<<dim3(E / 64, R / 64), blk, 0, stream>>>(ctxm, w_out, b_out, x, x, R, E, 256, 0);

  // convolution module
  gemm_bf16<<<dim3(1024 / 64, R / 64), blk, 0, stream>>>(x, w_pw1, b_pw1, nullptr, h1, R, 1024, E, 0);
  glu_kernel<<<(R * E) / 256, blk, 0, stream>>>(h1, glu);
  dwconv_kernel<<<(R * E) / 256, blk, 0, stream>>>(glu, w_dw, b_dw, dwo, L);
  gemm_bf16<<<dim3(E / 64, R / 64), blk, 0, stream>>>(dwo, w_pw2, b_pw2, x, x, R, E, E, 0);

  // final FFN + BasicNorm
  gemm_bf16<<<dim3(DFF / 64, R / 64), blk, 0, stream>>>(x, w_ff1, b_ff1, nullptr, h1, R, DFF, E, 1);
  gemm_bf16<<<dim3(E / 64, R / 64), blk, 0, stream>>>(h1, w_ff2, b_ff2, x, x, R, E, DFF, 0);
  basic_norm<<<R, blk, 0, stream>>>(x, neps, out_x, E);
}
